// OracleAD3D_CAttn_46505905881359
// MI455X (gfx1250) — matrix operand order repaired and checked by reading, not compiled, not run on hardware
//
#include <hip/hip_runtime.h>
#include <hip/hip_bf16.h>

typedef __attribute__((ext_vector_type(16))) __bf16 v16bf;
typedef __attribute__((ext_vector_type(8)))  float  v8f;

#define RB 272   // LDS row pitch in bytes for bf16 matrices: 128 elems + 8 pad

// ---- WMMA fragment loaders (bf16, 16x32 A / 32x16 B), per CDNA5 ISA layouts ----
static __device__ __forceinline__ v16bf frag_a(const char* base, int row0, int kByte, int lane) {
  const char* p = base + (size_t)(row0 + (lane & 15)) * RB + kByte + ((lane >> 4) << 4);
  union { v16bf v; uint4 q[2]; } u;
  u.q[0] = *(const uint4*)p;
  u.q[1] = *(const uint4*)(p + 32);
  return u.v;
}
static __device__ __forceinline__ v16bf frag_b(const char* base, int col0, int kByte, int lane) {
  const char* p = base + (size_t)(col0 + (lane & 15)) * RB + kByte + ((lane >> 4) << 4);
  union { v16bf v; uint4 q[2]; } u;
  u.q[0] = *(const uint4*)p;
  u.q[1] = *(const uint4*)(p + 32);
  return u.v;
}
static __device__ __forceinline__ v8f wmma_bf16(v16bf a, v16bf b, v8f c) {
  return __builtin_amdgcn_wmma_f32_16x16x32_bf16(false, a, false, b, (short)0, c, false, false);
}
static __device__ __forceinline__ float sigm(float x) { return 1.f / (1.f + __expf(-x)); }

// =====================================================================
// Encoder: per-variable LSTM (input size 1) + online temporal softmax pool
// grid = (32 vars, 4 batch chunks of 32), block = 256 threads (8 waves)
// =====================================================================
__global__ __launch_bounds__(256, 1) void enc_kernel(
    const float* __restrict__ X, const float* __restrict__ Wih,
    const float* __restrict__ Whh, const float* __restrict__ bvec,
    const float* __restrict__ pool_w, float* __restrict__ Cout)
{
  extern __shared__ char smem[];
  char* whh = smem;                          // 512*RB = 139264
  char* hbuf0 = whh + 512 * RB;              // 32*RB  = 8704
  char* hbuf1 = hbuf0 + 32 * RB;             // 8704
  float* S    = (float*)(hbuf1 + 32 * RB);   // 32*132
  float* xcol = S + 32 * 132;                // 32
  float* wihL = xcol + 32;                   // 512
  float* bL   = wihL + 512;                  // 512
  float* pwL  = bL + 512;                    // 128
  float* mL   = pwL + 128;                   // 32
  float* lL   = mL + 32;                     // 32
  float* scL  = lL + 32;                     // 32
  float* wL   = scL + 32;                    // 32
  float* red  = wL + 32;                     // 256

  const int n   = blockIdx.x;
  const int b0  = blockIdx.y * 32;
  const int tid = threadIdx.x;
  const int lane = tid & 31;
  const int wv  = tid >> 5;
  const int m0  = (wv >> 2) * 16;       // 0 or 16 (rows within 32-row chunk)
  const int d0t = (wv & 3) * 2;         // first of 2 d-tiles per wave

  const float* WhhN = Whh + (size_t)n * 512 * 128;
  for (int i = tid; i < 512 * 128; i += 256) {
    int r = i >> 7, c = i & 127;
    ((__bf16*)(whh + r * RB))[c] = (__bf16)WhhN[i];
  }
  for (int i = tid; i < 512; i += 256) { wihL[i] = Wih[n * 512 + i]; bL[i] = bvec[n * 512 + i]; }
  if (tid < 128) pwL[tid] = pool_w[n * 128 + tid];
  if (tid < 32) { mL[tid] = -1e30f; lL[tid] = 0.f; }
  for (int i = tid; i < 32 * 136; i += 256) ((__bf16*)hbuf0)[i] = (__bf16)0.f;
  for (int i = tid; i < 32 * 132; i += 256) S[i] = 0.f;

  float creg[2][8];
  #pragma unroll
  for (int a = 0; a < 2; ++a)
    #pragma unroll
    for (int j = 0; j < 8; ++j) creg[a][j] = 0.f;

  __syncthreads();

  for (int t = 0; t < 127; ++t) {
    char* hA = (t & 1) ? hbuf1 : hbuf0;
    char* hB = (t & 1) ? hbuf0 : hbuf1;
    if (tid < 32) xcol[tid] = X[((size_t)(b0 + tid) * 128 + t) * 32 + n];
    __syncthreads();

    v16bf afr[4];
    #pragma unroll
    for (int kt = 0; kt < 4; ++kt) afr[kt] = frag_a(hA, m0, kt * 64, lane);

    #pragma unroll
    for (int dd = 0; dd < 2; ++dd) {
      const int dt = d0t + dd;
      v8f acc[4] = {{}, {}, {}, {}};
      #pragma unroll
      for (int kt = 0; kt < 4; ++kt) {
        #pragma unroll
        for (int g = 0; g < 4; ++g) {
          v16bf bf = frag_b(whh, g * 128 + dt * 16, kt * 64, lane);
          acc[g] = wmma_bf16(afr[kt], bf, acc[g]);
        }
      }
      const int col = dt * 16 + (lane & 15);
      const int rb_ = m0 + ((lane >> 4) << 3);
      #pragma unroll
      for (int j = 0; j < 8; ++j) {
        const int b = rb_ + j;
        const float xv = xcol[b];
        float gi = acc[0][j] + xv * wihL[col]         + bL[col];
        float gf = acc[1][j] + xv * wihL[128 + col]   + bL[128 + col];
        float gg = acc[2][j] + xv * wihL[256 + col]   + bL[256 + col];
        float go = acc[3][j] + xv * wihL[384 + col]   + bL[384 + col];
        float c_ = sigm(gf) * creg[dd][j] + sigm(gi) * tanhf(gg);
        creg[dd][j] = c_;
        float hv = sigm(go) * tanhf(c_);
        ((__bf16*)(hB + b * RB))[col] = (__bf16)hv;
      }
    }
    __syncthreads();

    // ---- online softmax pooling over time ----
    {
      int b = tid >> 3, oct = tid & 7;
      const __bf16* hr = (const __bf16*)(hB + b * RB);
      float p = 0.f;
      #pragma unroll
      for (int k2 = 0; k2 < 16; ++k2) { int k = oct * 16 + k2; p += (float)hr[k] * pwL[k]; }
      red[tid] = p;
    }
    __syncthreads();
    if (tid < 32) {
      float s = 0.f;
      #pragma unroll
      for (int o = 0; o < 8; ++o) s += red[tid * 8 + o];
      float m = mL[tid], mn = fmaxf(m, s);
      float sc = __expf(m - mn), w = __expf(s - mn);
      lL[tid] = lL[tid] * sc + w;
      mL[tid] = mn; scL[tid] = sc; wL[tid] = w;
    }
    __syncthreads();
    for (int i = tid; i < 32 * 128; i += 256) {
      int b = i >> 7, k = i & 127;
      float hv = (float)((const __bf16*)(hB + b * RB))[k];
      S[b * 132 + k] = S[b * 132 + k] * scL[b] + wL[b] * hv;
    }
    __syncthreads();
  }
  for (int i = tid; i < 32 * 128; i += 256) {
    int b = i >> 7, k = i & 127;
    Cout[((size_t)(b0 + b) * 32 + n) * 128 + k] = S[b * 132 + k] / fmaxf(lL[b], 1e-30f);
  }
}

// =====================================================================
// Generic bf16 WMMA GEMM: Y[r,c] = sum_k A[r,k]*W[c,k] (+ epilogue)
// block: 128 rows x 64 cols; mode 0: Y=A@W^T+bias (ldy), mode 1: out-proj epilogue
// =====================================================================
__global__ __launch_bounds__(256, 1) void gemm_kernel(
    const float* __restrict__ A, const float* __restrict__ W, const float* __restrict__ bias,
    float* __restrict__ Yout, int ldy, int mode,
    const float* __restrict__ Cin, const float* __restrict__ wsum, float* __restrict__ Cst)
{
  extern __shared__ char smem[];
  char* As = smem;              // 128*RB
  char* Ws = As + 128 * RB;     // 64*RB
  const int tid = threadIdx.x, lane = tid & 31, wv = tid >> 5;
  const int r0 = blockIdx.x * 128, c0 = blockIdx.y * 64;
  for (int i = tid; i < 128 * 128; i += 256) {
    int r = i >> 7, c = i & 127;
    ((__bf16*)(As + r * RB))[c] = (__bf16)A[(size_t)(r0 + r) * 128 + c];
  }
  for (int i = tid; i < 64 * 128; i += 256) {
    int r = i >> 7, c = i & 127;
    ((__bf16*)(Ws + r * RB))[c] = (__bf16)W[(size_t)(c0 + r) * 128 + c];
  }
  __syncthreads();
  const int m0 = wv * 16;
  v16bf afr[4];
  #pragma unroll
  for (int kt = 0; kt < 4; ++kt) afr[kt] = frag_a(As, m0, kt * 64, lane);
  #pragma unroll
  for (int ct = 0; ct < 4; ++ct) {
    v8f acc = {};
    #pragma unroll
    for (int kt = 0; kt < 4; ++kt) {
      v16bf bf = frag_b(Ws, ct * 16, kt * 64, lane);
      acc = wmma_bf16(afr[kt], bf, acc);
    }
    const int c = c0 + ct * 16 + (lane & 15);
    const int rbase = r0 + m0 + ((lane >> 4) << 3);
    #pragma unroll
    for (int j = 0; j < 8; ++j) {
      const int r = rbase + j;
      if (mode == 0) {
        Yout[(size_t)r * ldy + c] = acc[j] + bias[c];
      } else {
        float ws = wsum[r >> 5];
        float outv = (acc[j] + bias[c] * ws) / fmaxf(ws, 1e-8f);
        Cst[(size_t)r * 128 + c] = Cin[(size_t)r * 128 + c] + outv;
      }
    }
  }
}

// =====================================================================
// Per-query-window lagged MHA, decay-weighted aggregation (pre-out-proj)
// grid = 128 (query windows), block = 256 (one thread per (head, query-var))
// =====================================================================
__global__ __launch_bounds__(256, 1) void attn_kernel(
    const float* __restrict__ Y, float* __restrict__ Oacc, float* __restrict__ wsumArr)
{
  extern __shared__ char smem[];
  float* q  = (float*)smem;    // 32*128
  float* kk = q + 32 * 128;    // 32*128
  float* vv = kk + 32 * 128;   // 32*128
  float* oa = vv + 32 * 128;   // 32*128
  const int qb = blockIdx.x, tid = threadIdx.x;
  for (int i = tid; i < 32 * 128; i += 256) {
    int r = i >> 7, c = i & 127;
    q[i] = Y[(size_t)(qb * 32 + r) * 384 + c];
    oa[i] = 0.f;
  }
  float wsum = 0.f;
  __syncthreads();
  const int h = tid >> 5, qi = tid & 31;
  float qreg[16];
  #pragma unroll
  for (int d = 0; d < 16; ++d) qreg[d] = q[qi * 128 + h * 16 + d];

  for (int tau = 1; tau <= 5; ++tau) {
    int kb = qb - tau;
    if (kb < 0) break;                 // uniform across block
    __syncthreads();
    for (int i = tid; i < 32 * 128; i += 256) {
      int r = i >> 7, c = i & 127;
      size_t base = (size_t)(kb * 32 + r) * 384;
      kk[i] = Y[base + 128 + c];
      vv[i] = Y[base + 256 + c];
    }
    __syncthreads();
    float w = __expf(-0.7f * (float)(tau - 1));
    wsum += w;
    float sc[32]; float mx = -1e30f;
    #pragma unroll
    for (int j = 0; j < 32; ++j) {
      float s = 0.f;
      #pragma unroll
      for (int d = 0; d < 16; ++d) s += qreg[d] * kk[j * 128 + h * 16 + d];
      s *= 0.25f; sc[j] = s; mx = fmaxf(mx, s);
    }
    float ssum = 0.f;
    #pragma unroll
    for (int j = 0; j < 32; ++j) { sc[j] = __expf(sc[j] - mx); ssum += sc[j]; }
    float scale = w / ssum;
    #pragma unroll
    for (int d = 0; d < 16; ++d) {
      float a = 0.f;
      #pragma unroll
      for (int j = 0; j < 32; ++j) a += sc[j] * vv[j * 128 + h * 16 + d];
      oa[qi * 128 + h * 16 + d] += a * scale;
    }
  }
  __syncthreads();
  for (int i = tid; i < 32 * 128; i += 256) Oacc[(size_t)qb * 32 * 128 + i] = oa[i];
  if (tid == 0) wsumArr[qb] = wsum;
}

// =====================================================================
// Decoder: per-variable LSTM (zero input), init from tanh(Linear(C*)),
// per-step scalar output. grid = (32 vars, 4 chunks), block = 256
// =====================================================================
__global__ __launch_bounds__(256, 1) void dec_kernel(
    const float* __restrict__ Cstar, const float* __restrict__ iHw, const float* __restrict__ iHb,
    const float* __restrict__ iCw, const float* __restrict__ iCb,
    const float* __restrict__ Whh, const float* __restrict__ bih, const float* __restrict__ bhh,
    const float* __restrict__ ow, const float* __restrict__ ob,
    float* __restrict__ recon, float* __restrict__ pred)
{
  extern __shared__ char smem[];
  char* whh = smem;                          // 512*RB
  char* hbuf0 = whh + 512 * RB;              // 32*RB
  char* hbuf1 = hbuf0 + 32 * RB;             // 32*RB
  float* cL   = (float*)(hbuf1 + 32 * RB);   // 32*132
  float* bsum = cL + 32 * 132;               // 512
  float* owL  = bsum + 512;                  // 128
  float* red  = owL + 128;                   // 256
  float* csL  = red + 256;                   // 32*128

  const int n = blockIdx.x, b0 = blockIdx.y * 32;
  const int tid = threadIdx.x, lane = tid & 31, wv = tid >> 5;
  const int m0 = (wv >> 2) * 16, d0t = (wv & 3) * 2;

  const float* WhhN = Whh + (size_t)n * 512 * 128;
  for (int i = tid; i < 512 * 128; i += 256) {
    int r = i >> 7, c = i & 127;
    ((__bf16*)(whh + r * RB))[c] = (__bf16)WhhN[i];
  }
  for (int i = tid; i < 512; i += 256) bsum[i] = bih[n * 512 + i] + bhh[n * 512 + i];
  if (tid < 128) owL[tid] = ow[n * 128 + tid];
  for (int i = tid; i < 32 * 128; i += 256) {
    int b = i >> 7, d = i & 127;
    csL[i] = Cstar[((size_t)(b0 + b) * 32 + n) * 128 + d];
  }
  __syncthreads();

  // init h0/c0 = tanh(C* @ W^T + b)  (one-time VALU GEMV)
  const float* iHwN = iHw + (size_t)n * 128 * 128;
  const float* iCwN = iCw + (size_t)n * 128 * 128;
  for (int i = tid; i < 32 * 128; i += 256) {
    int b = i >> 7, e = i & 127;
    const float* cs = csL + b * 128;
    const float* wh = iHwN + (size_t)e * 128;
    const float* wc = iCwN + (size_t)e * 128;
    float ah = 0.f, ac = 0.f;
    for (int d = 0; d < 128; ++d) { float x = cs[d]; ah += x * wh[d]; ac += x * wc[d]; }
    ((__bf16*)(hbuf0 + b * RB))[e] = (__bf16)tanhf(ah + iHb[n * 128 + e]);
    cL[b * 132 + e] = tanhf(ac + iCb[n * 128 + e]);
  }
  __syncthreads();

  const float obn = ob[n];
  for (int t = 0; t < 128; ++t) {
    char* hA = (t & 1) ? hbuf1 : hbuf0;
    char* hB = (t & 1) ? hbuf0 : hbuf1;

    v16bf afr[4];
    #pragma unroll
    for (int kt = 0; kt < 4; ++kt) afr[kt] = frag_a(hA, m0, kt * 64, lane);

    #pragma unroll
    for (int dd = 0; dd < 2; ++dd) {
      const int dt = d0t + dd;
      v8f acc[4] = {{}, {}, {}, {}};
      #pragma unroll
      for (int kt = 0; kt < 4; ++kt) {
        #pragma unroll
        for (int g = 0; g < 4; ++g) {
          v16bf bf = frag_b(whh, g * 128 + dt * 16, kt * 64, lane);
          acc[g] = wmma_bf16(afr[kt], bf, acc[g]);
        }
      }
      const int col = dt * 16 + (lane & 15);
      const int rb_ = m0 + ((lane >> 4) << 3);
      #pragma unroll
      for (int j = 0; j < 8; ++j) {
        const int b = rb_ + j;
        float gi = acc[0][j] + bsum[col];
        float gf = acc[1][j] + bsum[128 + col];
        float gg = acc[2][j] + bsum[256 + col];
        float go = acc[3][j] + bsum[384 + col];
        float c_ = sigm(gf) * cL[b * 132 + col] + sigm(gi) * tanhf(gg);
        cL[b * 132 + col] = c_;
        ((__bf16*)(hB + b * RB))[col] = (__bf16)(sigm(go) * tanhf(c_));
      }
    }
    __syncthreads();
    // output: o[t,b] = h . ow + ob
    {
      int b = tid >> 3, oct = tid & 7;
      const __bf16* hr = (const __bf16*)(hB + b * RB);
      float p = 0.f;
      #pragma unroll
      for (int k2 = 0; k2 < 16; ++k2) { int k = oct * 16 + k2; p += (float)hr[k] * owL[k]; }
      red[tid] = p;
    }
    __syncthreads();
    if (tid < 32) {
      float o = obn;
      #pragma unroll
      for (int x = 0; x < 8; ++x) o += red[tid * 8 + x];
      if (t < 127) recon[((size_t)(b0 + tid) * 127 + t) * 32 + n] = o;
      else         pred[(size_t)(b0 + tid) * 32 + n] = o;
    }
    __syncthreads();
  }
}

// =====================================================================
extern "C" void kernel_launch(void* const* d_in, const int* in_sizes, int n_in,
                              void* d_out, int out_size, void* d_ws, size_t ws_size,
                              hipStream_t stream) {
  (void)in_sizes; (void)n_in; (void)out_size; (void)ws_size;
  const float* X     = (const float*)d_in[0];
  const float* eWih  = (const float*)d_in[1];
  const float* eWhh  = (const float*)d_in[2];
  const float* eB    = (const float*)d_in[3];
  const float* poolw = (const float*)d_in[4];
  // d_in[5] pool_b: constant shift, cancels in time-softmax
  const float* inW   = (const float*)d_in[6];
  const float* inB   = (const float*)d_in[7];
  const float* outW  = (const float*)d_in[8];
  const float* outB  = (const float*)d_in[9];
  const float* iHw   = (const float*)d_in[10];
  const float* iHb   = (const float*)d_in[11];
  const float* iCw   = (const float*)d_in[12];
  const float* iCb   = (const float*)d_in[13];
  const float* dWhh  = (const float*)d_in[14];
  const float* dbih  = (const float*)d_in[15];
  const float* dbhh  = (const float*)d_in[16];
  const float* dow   = (const float*)d_in[17];
  const float* dob   = (const float*)d_in[18];

  float* outF  = (float*)d_out;
  float* recon = outF;             // (128,127,32)
  float* pred  = outF + 520192;    // (128,32)
  float* cstar = outF + 524288;    // (128,32,128)

  float* wsC = (float*)d_ws;       // C        (128,32,128)
  float* wsY = wsC + 524288;       // QKV      (4096,384)
  float* wsO = wsY + 1572864;      // Oacc     (4096,128)
  float* wsW = wsO + 524288;       // wsum     (128)

  constexpr size_t ENC_SMEM  = (size_t)512 * RB + 2 * 32 * RB + 5792 * sizeof(float);
  constexpr size_t DEC_SMEM  = (size_t)512 * RB + 2 * 32 * RB + 9216 * sizeof(float);
  constexpr size_t GEMM_SMEM = (size_t)(128 + 64) * RB;
  constexpr size_t ATTN_SMEM = (size_t)4 * 32 * 128 * sizeof(float);

  (void)hipFuncSetAttribute((const void*)enc_kernel,  hipFuncAttributeMaxDynamicSharedMemorySize, (int)ENC_SMEM);
  (void)hipFuncSetAttribute((const void*)dec_kernel,  hipFuncAttributeMaxDynamicSharedMemorySize, (int)DEC_SMEM);
  (void)hipFuncSetAttribute((const void*)gemm_kernel, hipFuncAttributeMaxDynamicSharedMemorySize, (int)GEMM_SMEM);
  (void)hipFuncSetAttribute((const void*)attn_kernel, hipFuncAttributeMaxDynamicSharedMemorySize, (int)ATTN_SMEM);

  enc_kernel<<<dim3(32, 4), 256, ENC_SMEM, stream>>>(X, eWih, eWhh, eB, poolw, wsC);
  gemm_kernel<<<dim3(32, 6), 256, GEMM_SMEM, stream>>>(wsC, inW, inB, wsY, 384, 0, nullptr, nullptr, nullptr);
  attn_kernel<<<dim3(128), 256, ATTN_SMEM, stream>>>(wsY, wsO, wsW);
  gemm_kernel<<<dim3(32, 2), 256, GEMM_SMEM, stream>>>(wsO, outW, outB, nullptr, 0, 1, wsC, wsW, cstar);
  dec_kernel<<<dim3(32, 4), 256, DEC_SMEM, stream>>>(cstar, iHw, iHb, iCw, iCb,
                                                     dWhh, dbih, dbhh, dow, dob, recon, pred);
}
